// SimpleGatedAttention_33457795236068
// MI455X (gfx1250) — matrix operand order repaired and checked by reading, not compiled, not run on hardware
//
#include <hip/hip_runtime.h>
#include <hip/hip_bf16.h>
#include <math.h>

// ---------------------------------------------------------------------------
// SimpleGatedAttention on MI455X (gfx1250)
//   out[G,1024] = segment_softmax( gelu(x@W_att + b_att) @ W_cls )-weighted
//                 segment-sum of x
// Dominant GEMM (131072x1024x256, 68.7 GFLOP) in BF16 WMMA / FP32 accum.
// W_att is pre-converted+transposed once to bf16 (512 KB, L2-resident) in a
// fragment-ready layout so B operands are two global_load_b128 per tile with
// zero per-step conversion; only the x tile is converted (once per element).
// ---------------------------------------------------------------------------

#define N_NODES 131072
#define IN_FEAT 1024
#define NHID    256
#define NSEG    64

typedef __attribute__((ext_vector_type(16))) __bf16 v16bf;
typedef __attribute__((ext_vector_type(8)))  float  v8f;

union Frag32 { uint4 u[2]; v16bf v; };

__device__ __forceinline__ unsigned short f2bf(float f) {
    unsigned u = __float_as_uint(f);
    u += 0x7FFFu + ((u >> 16) & 1u);        // round-to-nearest-even
    return (unsigned short)(u >> 16);
}

__device__ __forceinline__ unsigned pk_bf16(float a, float b) {
    __hip_bfloat162 h = __float22bfloat162_rn(make_float2(a, b));
    unsigned r;
    __builtin_memcpy(&r, &h, sizeof(r));     // not trivially copyable -> memcpy
    return r;
}

__device__ __forceinline__ unsigned ordkey(float f) {   // order-preserving uint
    unsigned u = __float_as_uint(f);
    return (u & 0x80000000u) ? ~u : (u | 0x80000000u);
}
__device__ __forceinline__ float orddecode(unsigned k) {
    unsigned u = (k & 0x80000000u) ? (k & 0x7FFFFFFFu) : ~k;
    return __uint_as_float(u);
}

// ---------------------------------------------------------------------------
// K0: one-off W_att f32[K=1024][N=256] -> bf16 W_T[N=256][K=1024]
// (fragment-ready: a WMMA lane's 16 K-values are contiguous)
// ---------------------------------------------------------------------------
__global__ __launch_bounds__(256) void convert_W_kernel(
    const float* __restrict__ W, unsigned short* __restrict__ Wt)
{
    const int k = blockIdx.x;      // 0..1023
    const int n = threadIdx.x;     // 0..255
    Wt[(size_t)n * IN_FEAT + k] = f2bf(W[(size_t)k * NHID + n]);
}

// ---------------------------------------------------------------------------
// K1: bottleneck GEMM (bf16 WMMA) + bias + exact GELU + W_cls dot -> A[n]
// Block = 256 thr = 8 waves; block tile = 32 rows x 256 cols; K step = 32.
// A tile staged through LDS (f32->bf16); B fragments direct from global W_T.
// ---------------------------------------------------------------------------
__global__ __launch_bounds__(256) void gemm_gelu_score(
    const float* __restrict__ x, const unsigned short* __restrict__ Wt,
    const float* __restrict__ b_att, const float* __restrict__ W_cls,
    float* __restrict__ A_out)
{
    // rows padded to 40 ushorts (80B) so fragment b128 loads stay 16B aligned
    __shared__ unsigned short Alds[32][40];     // 2.5 KB

    const int tid   = threadIdx.x;
    const int r0    = blockIdx.x * 32;
    const int lane  = tid & 31;
    const int l     = lane & 15;        // lane-in-half
    const int hi    = lane >> 4;        // which 16-lane half
    const int wave  = tid >> 5;
    const int waveM = wave >> 2;        // 0..1  -> 16-row group
    const int waveN = wave & 3;         // 0..3  -> 64-col group

    // B fragment base pointers (K-contiguous rows of W_T), hoisted
    const unsigned short* brow[4];
#pragma unroll
    for (int t = 0; t < 4; ++t)
        brow[t] = Wt + (size_t)(waveN * 64 + t * 16 + l) * IN_FEAT + hi * 16;

    const int ai = tid >> 3;            // staging row 0..31
    const int aj = (tid & 7) << 2;      // staging col 0,4,...,28
    const float* axp = x + (size_t)(r0 + ai) * IN_FEAT + aj;

    v8f acc[4] = {};

    for (int kk = 0; kk < IN_FEAT; kk += 32) {
        // ---- stage A tile: 32 rows x 32 K (f32 -> bf16, packed hw cvt) ----
        {
            const float4 v = *(const float4*)(axp + kk);
            *(uint2*)&Alds[ai][aj] =
                make_uint2(pk_bf16(v.x, v.y), pk_bf16(v.z, v.w));
        }
        __syncthreads();

        // A fragment: lanes 0-15 K 0..15, lanes 16-31 K 16..31 (contig)
        Frag32 af;
        const unsigned short* arow = &Alds[waveM * 16 + l][0];
        af.u[0] = *(const uint4*)(arow + hi * 16);
        af.u[1] = *(const uint4*)(arow + hi * 16 + 8);

#pragma unroll
        for (int t = 0; t < 4; ++t) {
            // B fragment: lanes 0-15 K 0..15, lanes 16-31 K 16..31 (contig)
            Frag32 bf;
            const uint4* bp = (const uint4*)(brow[t] + kk);
            bf.u[0] = bp[0];
            bf.u[1] = bp[1];
            acc[t] = __builtin_amdgcn_wmma_f32_16x16x32_bf16(
                false, af.v, false, bf.v, (short)0, acc[t], false, false);
        }
        __syncthreads();
    }

    // ---- epilogue: bias + exact GELU + W_cls dot, reduce over 256 cols ----
    float rowsum[8];
#pragma unroll
    for (int v = 0; v < 8; ++v) rowsum[v] = 0.0f;

#pragma unroll
    for (int t = 0; t < 4; ++t) {
        const int col = waveN * 64 + t * 16 + l;
        const float bias = b_att[col];
        const float wc   = W_cls[col];
#pragma unroll
        for (int v = 0; v < 8; ++v) {
            float z = acc[t][v] + bias;
            float g = 0.5f * z * (1.0f + erff(z * 0.70710678118654752f));
            rowsum[v] += g * wc;
        }
    }
    // C/D layout: VGPR v -> row (v + hi*8); reduce across the 16 lanes of a half
#pragma unroll
    for (int v = 0; v < 8; ++v) {
        float s = rowsum[v];
        s += __shfl_xor(s, 1, 32);
        s += __shfl_xor(s, 2, 32);
        s += __shfl_xor(s, 4, 32);
        s += __shfl_xor(s, 8, 32);
        if (l == 0)
            atomicAdd(&A_out[r0 + waveM * 16 + hi * 8 + v], s);
    }
}

// ---------------------------------------------------------------------------
// K2: per-segment max via order-preserving uint atomicMax (memset-0 identity)
// ---------------------------------------------------------------------------
__global__ __launch_bounds__(256) void seg_max_kernel(
    const float* __restrict__ A, const int* __restrict__ seg,
    unsigned* __restrict__ segmax_u)
{
    const int n    = blockIdx.x * 256 + threadIdx.x;
    const int lane = threadIdx.x & 31;
    const int s    = seg[n];
    unsigned key   = ordkey(A[n]);
    const int s0   = __shfl(s, 0, 32);
    if (__all(s == s0)) {                    // wave-uniform segment fast path
#pragma unroll
        for (int off = 16; off; off >>= 1) {
            unsigned o = __shfl_xor(key, off, 32);
            key = (o > key) ? o : key;
        }
        if (lane == 0) atomicMax(&segmax_u[s0], key);
    } else {
        atomicMax(&segmax_u[s], key);
    }
}

// ---------------------------------------------------------------------------
// K3: e = exp(A - segmax) ; segsum += e
// ---------------------------------------------------------------------------
__global__ __launch_bounds__(256) void seg_exp_sum_kernel(
    const float* __restrict__ A, const int* __restrict__ seg,
    const unsigned* __restrict__ segmax_u,
    float* __restrict__ e, float* __restrict__ segsum)
{
    const int n    = blockIdx.x * 256 + threadIdx.x;
    const int lane = threadIdx.x & 31;
    const int s    = seg[n];
    const float m  = orddecode(segmax_u[s]);
    const float v  = expf(A[n] - m);
    e[n] = v;
    const int s0 = __shfl(s, 0, 32);
    if (__all(s == s0)) {
        float t = v;
#pragma unroll
        for (int off = 16; off; off >>= 1) t += __shfl_xor(t, off, 32);
        if (lane == 0) atomicAdd(&segsum[s0], t);
    } else {
        atomicAdd(&segsum[s], v);
    }
}

// ---------------------------------------------------------------------------
// K4: w = e / segsum[seg]
// ---------------------------------------------------------------------------
__global__ __launch_bounds__(256) void compute_w_kernel(
    const float* __restrict__ e, const int* __restrict__ seg,
    const float* __restrict__ segsum, float* __restrict__ w)
{
    const int n = blockIdx.x * 256 + threadIdx.x;
    w[n] = e[n] / segsum[seg[n]];
}

// ---------------------------------------------------------------------------
// K5: out[seg] += w[n] * x[n,:]  (segment_ids sorted -> register accumulation,
// atomic flush only on segment boundary). Thread owns 4 features, block owns
// the whole 1024-feature row over a 256-node chunk.
// ---------------------------------------------------------------------------
__global__ __launch_bounds__(256) void weighted_segsum_kernel(
    const float* __restrict__ x, const float* __restrict__ w,
    const int* __restrict__ seg, float* __restrict__ out)
{
    const int f  = threadIdx.x * 4;
    const int n0 = blockIdx.x * 256;
    float4 acc = make_float4(0.f, 0.f, 0.f, 0.f);
    int cur = seg[n0];

    for (int i = 0; i < 256; ++i) {
        const int n = n0 + i;
        __builtin_prefetch(x + (size_t)(n + 8) * IN_FEAT + f, 0, 0);
        const int s = seg[n];
        if (s != cur) {
            float* o = out + (size_t)cur * IN_FEAT + f;
            atomicAdd(o + 0, acc.x); atomicAdd(o + 1, acc.y);
            atomicAdd(o + 2, acc.z); atomicAdd(o + 3, acc.w);
            acc = make_float4(0.f, 0.f, 0.f, 0.f);
            cur = s;
        }
        const float wn = w[n];
        const float4 xv = *(const float4*)(x + (size_t)n * IN_FEAT + f);
        acc.x += wn * xv.x; acc.y += wn * xv.y;
        acc.z += wn * xv.z; acc.w += wn * xv.w;
    }
    float* o = out + (size_t)cur * IN_FEAT + f;
    atomicAdd(o + 0, acc.x); atomicAdd(o + 1, acc.y);
    atomicAdd(o + 2, acc.z); atomicAdd(o + 3, acc.w);
}

// ---------------------------------------------------------------------------
extern "C" void kernel_launch(void* const* d_in, const int* in_sizes, int n_in,
                              void* d_out, int out_size, void* d_ws, size_t ws_size,
                              hipStream_t stream) {
    const float* x     = (const float*)d_in[0];
    const float* W_att = (const float*)d_in[1];
    const float* b_att = (const float*)d_in[2];
    const float* W_cls = (const float*)d_in[3];
    // d_in[4] = b_cls: constant shift of every node's score -> softmax
    // invariant, so it is (correctly) unused.
    const int* seg     = (const int*)d_in[5];
    float* out         = (float*)d_out;

    // workspace: W_T bf16[256][1024] | A[N] | e[N] | w[N] | segmax_u | segsum
    unsigned short* Wt = (unsigned short*)d_ws;
    float*    A_ws     = (float*)((char*)d_ws + (size_t)NHID * IN_FEAT * 2);
    float*    e_ws     = A_ws + N_NODES;
    float*    w_ws     = e_ws + N_NODES;
    unsigned* segmax_u = (unsigned*)(w_ws + N_NODES);
    float*    segsum   = (float*)(segmax_u + NSEG);
    (void)in_sizes; (void)n_in; (void)ws_size;

    (void)hipMemsetAsync(A_ws, 0, N_NODES * sizeof(float), stream);
    (void)hipMemsetAsync(segmax_u, 0, 2 * NSEG * sizeof(unsigned), stream);
    (void)hipMemsetAsync(d_out, 0, (size_t)out_size * sizeof(float), stream);

    convert_W_kernel<<<IN_FEAT, NHID, 0, stream>>>(W_att, Wt);
    gemm_gelu_score<<<N_NODES / 32, 256, 0, stream>>>(x, Wt, b_att, W_cls, A_ws);
    seg_max_kernel<<<N_NODES / 256, 256, 0, stream>>>(A_ws, seg, segmax_u);
    seg_exp_sum_kernel<<<N_NODES / 256, 256, 0, stream>>>(A_ws, seg, segmax_u,
                                                          e_ws, segsum);
    compute_w_kernel<<<N_NODES / 256, 256, 0, stream>>>(e_ws, seg, segsum, w_ws);
    weighted_segsum_kernel<<<N_NODES / 256, 256, 0, stream>>>(x, w_ws, seg, out);
}
